// SAST_block_9698036155059
// MI455X (gfx1250) — matrix operand order repaired and checked by reading, not compiled, not run on hardware
//
#include <hip/hip_runtime.h>
#include <hip/hip_bf16.h>

// ---- problem constants (fixed by reference) ----
#define NWIN  2048
#define PP    64
#define CC    256
#define MSEL  1536
#define KEEPT 48
#define REALT 40
#define HH    8
#define DHH   32

typedef __attribute__((ext_vector_type(16))) _Float16 v16h;
typedef __attribute__((ext_vector_type(8)))  float    v8f;

#define WMMA_F32F16(a,b,c) \
  __builtin_amdgcn_wmma_f32_16x16x32_f16(false,(a),false,(b),(short)0,(c),false,false)

// ---------------- WMMA fragment loaders (wave32; contiguous k order k = 16h + s, same for A and B) ----------------
// A: 16x32 f16, row-major tile in LDS (leading dim ld); all 16 halves contiguous
// per lane-half: lanes 0-15 rows, K 0..15; lanes 16-31 K 16..31.
static __device__ __forceinline__ v16h frag_a_lds(const _Float16* p, int ld) {
  int lane = threadIdx.x & 31;
  const _Float16* r = p + (lane & 15) * ld + ((lane & 16) ? 16 : 0);
  v16h a;
#pragma unroll
  for (int i = 0; i < 16; ++i) a[i] = r[i];
  return a;
}
// B: 32x16 f16 from an N-major (transposed) source: element (k,n) at p[n*ld + k].
// lanes 0-15 cols with K 0..15, lanes 16-31 cols with K 16..31 -> contiguous reads.
static __device__ __forceinline__ v16h frag_bt_lds(const _Float16* p, int ld) {
  int lane = threadIdx.x & 31;
  const _Float16* r = p + (lane & 15) * ld + ((lane & 16) ? 16 : 0);
  v16h b;
#pragma unroll
  for (int i = 0; i < 16; ++i) b[i] = r[i];
  return b;
}

static __device__ __forceinline__ float gelu_exact(float x) {
  return 0.5f * x * (1.0f + erff(x * 0.7071067811865476f));
}

// ---------------- LayerNorm: one wave (32 lanes) per 256-wide row ----------------
__global__ __launch_bounds__(256) void ln1_kernel(const float* __restrict__ in,
                                                  float* __restrict__ out,
                                                  const float* __restrict__ g,
                                                  const float* __restrict__ b,
                                                  int nrows) {
  int wave = threadIdx.x >> 5, lane = threadIdx.x & 31;
  int row = blockIdx.x * 8 + wave;
  if (row >= nrows) return;
  const float* src = in + (size_t)row * CC;
  float v[8], s = 0.f;
#pragma unroll
  for (int i = 0; i < 8; ++i) { v[i] = src[lane + 32 * i]; s += v[i]; }
#pragma unroll
  for (int off = 16; off > 0; off >>= 1) s += __shfl_xor(s, off, 32);
  float mu = s * (1.f / CC), vs = 0.f;
#pragma unroll
  for (int i = 0; i < 8; ++i) { float d = v[i] - mu; vs += d * d; }
#pragma unroll
  for (int off = 16; off > 0; off >>= 1) vs += __shfl_xor(vs, off, 32);
  float rstd = rsqrtf(vs * (1.f / CC) + 1e-5f);
  float* dst = out + (size_t)row * CC;
#pragma unroll
  for (int i = 0; i < 8; ++i) {
    int c = lane + 32 * i;
    dst[c] = (v[i] - mu) * rstd * g[c] + b[c];
  }
}

// LN2 applied in place on XX rows selected by asy_index; also saves shortcut.
__global__ __launch_bounds__(256) void ln2_kernel(float* __restrict__ XX,
                                                  const int* __restrict__ asy,
                                                  const float* __restrict__ g,
                                                  const float* __restrict__ b,
                                                  float* __restrict__ shortcut,
                                                  int nrows) {
  int wave = threadIdx.x >> 5, lane = threadIdx.x & 31;
  int i = blockIdx.x * 8 + wave;
  if (i >= nrows) return;
  float* src = XX + (size_t)asy[i] * CC;
  float v[8], s = 0.f;
#pragma unroll
  for (int k = 0; k < 8; ++k) { v[k] = src[lane + 32 * k]; s += v[k]; }
#pragma unroll
  for (int off = 16; off > 0; off >>= 1) s += __shfl_xor(s, off, 32);
  float mu = s * (1.f / CC), vs = 0.f;
#pragma unroll
  for (int k = 0; k < 8; ++k) { float d = v[k] - mu; vs += d * d; }
#pragma unroll
  for (int off = 16; off > 0; off >>= 1) vs += __shfl_xor(vs, off, 32);
  float rstd = rsqrtf(vs * (1.f / CC) + 1e-5f);
  float* sc = shortcut + (size_t)i * CC;
#pragma unroll
  for (int k = 0; k < 8; ++k) {
    int c = lane + 32 * k;
    float y = (v[k] - mu) * rstd * g[c] + b[c];
    src[c] = y; sc[c] = y;
  }
}

// ---------------- gather / pack / elementwise kernels ----------------
__global__ __launch_bounds__(256) void gather_windows_kernel(const float* __restrict__ X,
                                                             const int* __restrict__ iw,
                                                             float* __restrict__ XX, int nrows) {
  int wave = threadIdx.x >> 5, lane = threadIdx.x & 31;
  int t = blockIdx.x * 8 + wave;
  if (t >= nrows) return;
  int w = t >> 6, p = t & 63;
  const float4* src = (const float4*)(X + ((size_t)iw[w] * PP + p) * CC);
  float4* dst = (float4*)(XX + (size_t)t * CC);
  dst[lane] = src[lane];
  dst[lane + 32] = src[lane + 32];
}

__global__ __launch_bounds__(256) void pack_tokens_f16_kernel(const float* __restrict__ XX,
                                                              const int* __restrict__ itok,
                                                              _Float16* __restrict__ xt, int nrows) {
  int wave = threadIdx.x >> 5, lane = threadIdx.x & 31;
  int i = blockIdx.x * 8 + wave;
  if (i >= nrows) return;
  const float* src = XX + (size_t)itok[i] * CC;
  _Float16* dst = xt + (size_t)i * CC;
#pragma unroll
  for (int k = 0; k < 8; ++k) { int c = lane + 32 * k; dst[c] = (_Float16)src[c]; }
}

// transpose + downcast: in is KxN row-major f32; out is NxK row-major f16
__global__ void cast_transpose_f16_kernel(const float* __restrict__ in,
                                          _Float16* __restrict__ out, int K, int N) {
  int i = blockIdx.x * blockDim.x + threadIdx.x;
  if (i >= K * N) return;
  int n = i / K, k = i - n * K;
  out[i] = (_Float16)in[(size_t)k * N + n];
}

// xa = shortcut + XX[asy]*ls1  -> xa_f32 and xa_f16
__global__ void make_xa_kernel(const float* __restrict__ shortcut, const float* __restrict__ XX,
                               const int* __restrict__ asy, const float* __restrict__ ls1,
                               float* __restrict__ xaf, _Float16* __restrict__ xah, int n) {
  int i = blockIdx.x * blockDim.x + threadIdx.x;
  if (i >= n) return;
  int row = i >> 8, c = i & 255;
  float v = shortcut[i] + XX[(size_t)asy[row] * CC + c] * ls1[c];
  xaf[i] = v; xah[i] = (_Float16)v;
}

// restore padded rows of XX from the (still-x1) output buffer
__global__ void restore_pad_kernel(float* __restrict__ XX, const float* __restrict__ Xout,
                                   const int* __restrict__ pad, const int* __restrict__ iw, int n) {
  int i = blockIdx.x * blockDim.x + threadIdx.x;
  if (i >= n) return;
  int pr = i >> 8, c = i & 255;
  int t = pad[pr];
  int w = t >> 6, p = t & 63;
  XX[(size_t)t * CC + c] = Xout[((size_t)iw[w] * PP + p) * CC + c];
}

// final window scatter XX -> X
__global__ void scatter_windows_kernel(const float* __restrict__ XX, float* __restrict__ Xout,
                                       const int* __restrict__ iw, int n) {
  int i = blockIdx.x * blockDim.x + threadIdx.x;
  if (i >= n) return;
  int t = i >> 8, c = i & 255;
  int w = t >> 6, p = t & 63;
  Xout[((size_t)iw[w] * PP + p) * CC + c] = XX[i];
}

// ---------------- tiled WMMA GEMM, 64x128 block tile, 256 threads / 8 waves ----------------
// B is pre-transposed (NxK row-major f16) so all LDS fragment reads are contiguous b128.
// Each wave computes 16 rows x 64 cols = 4 WMMA per 32-deep K step.
// EPI 0: out_h = f16(acc + bias)
// EPI 1: out_h = f16(gelu(acc + bias))
// EPI 2: out_f[rowmap[row]] = acc + bias                     (proj scatter into XX)
// EPI 3: out_f[rowmap[row]] = resid[row] + (acc + bias)*ls   (MLP2 scatter into XX)
template <int EPI>
__global__ __launch_bounds__(256) void gemm64x128_kernel(const _Float16* __restrict__ A,
                                                         const _Float16* __restrict__ BT,
                                                         const float* __restrict__ bias,
                                                         int K, int N,
                                                         _Float16* __restrict__ outh,
                                                         float* __restrict__ outf, int ldo,
                                                         const int* __restrict__ rowmap,
                                                         const float* __restrict__ ls,
                                                         const float* __restrict__ resid) {
  __shared__ __align__(16) _Float16 sA[64 * 32];    // 64 rows x 32 k
  __shared__ __align__(16) _Float16 sBt[128 * 32];  // 128 cols x 32 k (N-major)
  int row0 = blockIdx.x * 64;
  int col0 = blockIdx.y * 128;
  int t = threadIdx.x;
  int wave = t >> 5, lane = t & 31;
  int wrow = wave & 3;   // 4 row-waves x 16 rows
  int wcol = wave >> 2;  // 2 col-waves x 64 cols
  int ar = t >> 2, ac = (t & 3) * 8;         // A loader: 64x32
  int bn = t >> 1, bk = (t & 1) * 16;        // B loader: 128x32, 2 float4 per thread

  v8f acc[4] = {{}, {}, {}, {}};
  for (int k0 = 0; k0 < K; k0 += 32) {
    *(float4*)&sA[ar * 32 + ac] = *(const float4*)&A[(size_t)(row0 + ar) * K + k0 + ac];
    *(float4*)&sBt[bn * 32 + bk]     = *(const float4*)&BT[(size_t)(col0 + bn) * K + k0 + bk];
    *(float4*)&sBt[bn * 32 + bk + 8] = *(const float4*)&BT[(size_t)(col0 + bn) * K + k0 + bk + 8];
    if (k0 + 32 < K) {
      __builtin_prefetch(&A[(size_t)(row0 + ar) * K + k0 + 32 + ac], 0, 1);
      __builtin_prefetch(&BT[(size_t)(col0 + bn) * K + k0 + 32 + bk], 0, 1);
    }
    __syncthreads();
    v16h a = frag_a_lds(&sA[(wrow * 16) * 32], 32);
#pragma unroll
    for (int ct = 0; ct < 4; ++ct) {
      v16h b = frag_bt_lds(&sBt[(wcol * 64 + ct * 16) * 32], 32);
      acc[ct] = WMMA_F32F16(a, b, acc[ct]);
    }
    __syncthreads();
  }

  int rhi = (lane & 16) ? 8 : 0;
#pragma unroll
  for (int ct = 0; ct < 4; ++ct) {
    int gcol = col0 + wcol * 64 + ct * 16 + (lane & 15);
    float bv = bias[gcol];
#pragma unroll
    for (int r = 0; r < 8; ++r) {
      int grow = row0 + wrow * 16 + rhi + r;
      float v = acc[ct][r] + bv;
      if (EPI == 0) {
        outh[(size_t)grow * ldo + gcol] = (_Float16)v;
      } else if (EPI == 1) {
        outh[(size_t)grow * ldo + gcol] = (_Float16)gelu_exact(v);
      } else if (EPI == 2) {
        outf[(size_t)rowmap[grow] * ldo + gcol] = v;
      } else {
        outf[(size_t)rowmap[grow] * ldo + gcol] =
            resid[(size_t)grow * ldo + gcol] + v * ls[gcol];
      }
    }
  }
}

// ---------------- fused per-(window,head) attention; one wave per head ----------------
// qkv rows: (m*KEEP + tok) * 768, head h occupies cols [h*96, h*96+96): 32 q | 32 k | 32 v
__global__ __launch_bounds__(64) void attn_kernel(const _Float16* __restrict__ qkv,
                                                  _Float16* __restrict__ outh) {
  __shared__ __align__(16) _Float16 s_qk[2][48 * 64];  // q (cols 0..31) | k (cols 32..63)
  __shared__ __align__(16) _Float16 s_vt[2][32 * 64];  // V^T: [dim][key], keys 48..63 zero
  __shared__ float s_S[2][48 * 48];
  __shared__ __align__(16) _Float16 s_P[2][48 * 64];   // probs, cols 48..63 zero

  int wave = threadIdx.x >> 5, lane = threadIdx.x & 31;
  int m = blockIdx.x;
  int h = blockIdx.y * 2 + wave;
  _Float16* QK = s_qk[wave];
  _Float16* VT = s_vt[wave];
  float*    S  = s_S[wave];
  _Float16* P  = s_P[wave];

  // stage q|k rows (48 x 64 halves, contiguous copies)
  for (int idx = lane; idx < 48 * 8; idx += 32) {
    int r = idx >> 3, c8 = (idx & 7) * 8;
    *(float4*)&QK[r * 64 + c8] =
        *(const float4*)&qkv[((size_t)m * KEEPT + r) * (3 * CC) + h * 96 + c8];
  }
  // stage V transposed: VT[d*64 + key]
  for (int idx = lane; idx < 48 * 4; idx += 32) {
    int r = idx >> 2, d8 = (idx & 3) * 8;
    float4 t4 = *(const float4*)&qkv[((size_t)m * KEEPT + r) * (3 * CC) + h * 96 + 64 + d8];
    const _Float16* tv = (const _Float16*)&t4;
#pragma unroll
    for (int j = 0; j < 8; ++j) VT[(d8 + j) * 64 + r] = tv[j];
  }
  // zero the key-padding region of V^T (keys 48..63)
  float4 z4 = {0.f, 0.f, 0.f, 0.f};
  for (int idx = lane; idx < 32 * 2; idx += 32) {
    int d = idx >> 1, r8 = 48 + (idx & 1) * 8;
    *(float4*)&VT[d * 64 + r8] = z4;
  }

  // scores: S = (q @ k^T) * scale ; 3x3 tiles, one WMMA each (K = DH = 32)
  const float scale = 0.1767766952966369f;  // 32^-0.5
  int rhi = (lane & 16) ? 8 : 0;
#pragma unroll
  for (int qt = 0; qt < 3; ++qt) {
    v16h aq = frag_a_lds(QK + qt * 16 * 64, 64);
#pragma unroll
    for (int kt = 0; kt < 3; ++kt) {
      v16h bk = frag_bt_lds(QK + kt * 16 * 64 + 32, 64);  // B(k,n) = K[kt*16+n][k]
      v8f c = {};
      c = WMMA_F32F16(aq, bk, c);
      int col = kt * 16 + (lane & 15);
#pragma unroll
      for (int r = 0; r < 8; ++r) S[(qt * 16 + rhi + r) * 48 + col] = c[r] * scale;
    }
  }

  // masked softmax: keys j >= REALT are forced to -10000 (padding rows)
  for (int row = lane; row < KEEPT; row += 32) {
    const float* sr = S + row * 48;
    float mx = -10000.0f;
    for (int j = 0; j < REALT; ++j) mx = fmaxf(mx, sr[j]);
    float sum = 0.f;
    for (int j = 0; j < REALT; ++j) sum += __expf(sr[j] - mx);
    float pe = __expf(-10000.0f - mx);
    sum += (float)(KEEPT - REALT) * pe;
    float inv = 1.0f / sum;
    _Float16* pr = P + row * 64;
    for (int j = 0; j < REALT; ++j) pr[j] = (_Float16)(__expf(sr[j] - mx) * inv);
    _Float16 pv = (_Float16)(pe * inv);
    for (int j = REALT; j < KEEPT; ++j) pr[j] = pv;
    for (int j = KEEPT; j < 64; ++j) pr[j] = (_Float16)0.f;
  }

  // out = P @ V : K padded 48->64, two WMMA K-steps per tile; DH=32 -> 2 col tiles.
  // V fragments are loop-invariant: hoist (4 frags, contiguous b128 reads from V^T).
  v16h vb[2][2];
#pragma unroll
  for (int nt = 0; nt < 2; ++nt) {
    vb[nt][0] = frag_bt_lds(VT + nt * 16 * 64, 64);       // keys 0..31
    vb[nt][1] = frag_bt_lds(VT + nt * 16 * 64 + 32, 64);  // keys 32..63 (48+ zero)
  }
#pragma unroll
  for (int qt = 0; qt < 3; ++qt) {
    v16h a0 = frag_a_lds(P + qt * 16 * 64, 64);
    v16h a1 = frag_a_lds(P + qt * 16 * 64 + 32, 64);
#pragma unroll
    for (int nt = 0; nt < 2; ++nt) {
      v8f c = {};
      c = WMMA_F32F16(a0, vb[nt][0], c);
      c = WMMA_F32F16(a1, vb[nt][1], c);
      int gcol = h * DHH + nt * 16 + (lane & 15);
#pragma unroll
      for (int r = 0; r < 8; ++r) {
        int grow = m * KEEPT + qt * 16 + rhi + r;
        outh[(size_t)grow * CC + gcol] = (_Float16)c[r];
      }
    }
  }
}

// ---------------- launch ----------------
extern "C" void kernel_launch(void* const* d_in, const int* in_sizes, int n_in,
                              void* d_out, int out_size, void* d_ws, size_t ws_size,
                              hipStream_t stream) {
  const float* x       = (const float*)d_in[0];
  const float* qkv_w   = (const float*)d_in[1];
  const float* qkv_b   = (const float*)d_in[2];
  const float* proj_w  = (const float*)d_in[3];
  const float* proj_b  = (const float*)d_in[4];
  const float* n1g     = (const float*)d_in[5];
  const float* n1b     = (const float*)d_in[6];
  const float* n2g     = (const float*)d_in[7];
  const float* n2b     = (const float*)d_in[8];
  const float* ls1     = (const float*)d_in[9];
  const float* ls2     = (const float*)d_in[10];
  const float* mlp_w1  = (const float*)d_in[11];
  const float* mlp_b1  = (const float*)d_in[12];
  const float* mlp_w2  = (const float*)d_in[13];
  const float* mlp_b2  = (const float*)d_in[14];
  const int* index_window = (const int*)d_in[15];
  const int* index_token  = (const int*)d_in[16];
  const int* padding_index= (const int*)d_in[17];
  const int* asy_index    = (const int*)d_in[18];
  float* X = (float*)d_out;

  // workspace layout (all offsets 256B aligned)
  char* ws = (char*)d_ws;
  float*    XX     = (float*)(ws + 0);                        // 98304*256 f32
  float*    shortc = (float*)(ws + 100663296);                // 61440*256 f32
  float*    xa_f   = (float*)(ws + 163577856);                // 61440*256 f32
  _Float16* xt_h   = (_Float16*)(ws + 226492416);             // 73728*256 f16 (reused as attn out)
  _Float16* qkv_h  = (_Float16*)(ws + 264241152);             // 73728*768 f16
  _Float16* xa_h   = (_Float16*)(ws + 377487360);             // 61440*256 f16
  _Float16* hdn_h  = (_Float16*)(ws + 408944640);             // 61440*1024 f16
  _Float16* wqkvT  = (_Float16*)(ws + 534773760);             // 768x256 f16 (transposed)
  _Float16* wprojT = (_Float16*)(ws + 535166976);             // 256x256 f16 (transposed)
  _Float16* w1T    = (_Float16*)(ws + 535298048);             // 1024x256 f16 (transposed)
  _Float16* w2T    = (_Float16*)(ws + 535822336);             // 256x1024 f16 (transposed)
  _Float16* attn_h = xt_h;  // xt dead after QKV GEMM

  const int rowsAll  = NWIN * PP;          // 131072
  const int rowsSel  = MSEL * PP;          // 98304
  const int rowsKeep = MSEL * KEEPT;       // 73728
  const int rowsReal = MSEL * REALT;       // 61440

  // 1. LN1 over everything -> X (output base)
  ln1_kernel<<<rowsAll / 8, 256, 0, stream>>>(x, X, n1g, n1b, rowsAll);

  // 2. weight transpose + f16 packs (BT layout: N x K)
  cast_transpose_f16_kernel<<<(CC * 3 * CC + 255) / 256, 256, 0, stream>>>(qkv_w, wqkvT, CC, 3 * CC);
  cast_transpose_f16_kernel<<<(CC * CC + 255) / 256, 256, 0, stream>>>(proj_w, wprojT, CC, CC);
  cast_transpose_f16_kernel<<<(CC * 4 * CC + 255) / 256, 256, 0, stream>>>(mlp_w1, w1T, CC, 4 * CC);
  cast_transpose_f16_kernel<<<(4 * CC * CC + 255) / 256, 256, 0, stream>>>(mlp_w2, w2T, 4 * CC, CC);

  // 3. gather selected windows -> XX
  gather_windows_kernel<<<rowsSel / 8, 256, 0, stream>>>(X, index_window, XX, rowsSel);

  // 4. LN2 in place on real rows + shortcut
  ln2_kernel<<<rowsReal / 8, 256, 0, stream>>>(XX, asy_index, n2g, n2b, shortc, rowsReal);

  // 5. pack kept tokens to f16
  pack_tokens_f16_kernel<<<rowsKeep / 8, 256, 0, stream>>>(XX, index_token, xt_h, rowsKeep);

  // 6. QKV GEMM: (73728x256) x (256x768) + b -> f16
  gemm64x128_kernel<0><<<dim3(rowsKeep / 64, (3 * CC) / 128), 256, 0, stream>>>(
      xt_h, wqkvT, qkv_b, CC, 3 * CC, qkv_h, nullptr, 3 * CC, nullptr, nullptr, nullptr);

  // 7. fused masked attention, one wave per (window, head)
  attn_kernel<<<dim3(MSEL, HH / 2), 64, 0, stream>>>(qkv_h, attn_h);

  // 8. proj GEMM with scatter into XX: (73728x256) x (256x256) + b
  gemm64x128_kernel<2><<<dim3(rowsKeep / 64, CC / 128), 256, 0, stream>>>(
      attn_h, wprojT, proj_b, CC, CC, nullptr, XX, CC, index_token, nullptr, nullptr);

  // 9. xa = shortcut + XX[asy]*ls1
  make_xa_kernel<<<(rowsReal * CC + 255) / 256, 256, 0, stream>>>(
      shortc, XX, asy_index, ls1, xa_f, xa_h, rowsReal * CC);

  // 10. MLP fc1 + exact GELU: (61440x256) x (256x1024)
  gemm64x128_kernel<1><<<dim3(rowsReal / 64, (4 * CC) / 128), 256, 0, stream>>>(
      xa_h, w1T, mlp_b1, CC, 4 * CC, hdn_h, nullptr, 4 * CC, nullptr, nullptr, nullptr);

  // 11. MLP fc2 + LayerScale residual, scatter into XX: (61440x1024) x (1024x256)
  gemm64x128_kernel<3><<<dim3(rowsReal / 64, CC / 128), 256, 0, stream>>>(
      hdn_h, w2T, mlp_b2, 4 * CC, CC, nullptr, XX, CC, asy_index, ls2, xa_f);

  // 12. restore padded rows of XX from (still-x1) X
  restore_pad_kernel<<<(MSEL * (KEEPT - REALT) * CC + 255) / 256, 256, 0, stream>>>(
      XX, X, padding_index, index_window, MSEL * (KEEPT - REALT) * CC);

  // 13. scatter windows back into X
  scatter_windows_kernel<<<(rowsSel * CC + 255) / 256, 256, 0, stream>>>(
      XX, X, index_window, rowsSel * CC);
}
